// MyNewAttention_18004502905190
// MI455X (gfx1250) — matrix operand order repaired and checked by reading, not compiled, not run on hardware
//
#include <hip/hip_runtime.h>
#include <hip/hip_bf16.h>

typedef unsigned short ushort_t;
typedef __bf16 v16bf __attribute__((ext_vector_type(16)));
typedef float  v8f   __attribute__((ext_vector_type(8)));

#define B_   64
#define L_   80
#define K_   36
#define DV_  256
#define DQ_  512
#define H_   512
#define KP_  48        // padded detection rows (3 M-tiles)
#define NEGINF_ (-1e9f)

// LDS pitches (in halves): Dv rows padded 256->264, Dq rows padded 512->520.
// 264*2=528B (16B aligned, 528/4=132 dwords == 4 mod 64 -> conflict-free b128)
#define PV_  264
#define PQ_  520

// workspace layout (bytes)
#define OFF_SCAL  0u          // 4 floats
#define OFF_WL    256u        // 512 floats
#define OFF_WV    4096u       // 512x256 bf16, fragment-tiled (262144 B)
#define OFF_WQ    266240u     // 512x512 bf16, fragment-tiled (524288 B)
#define OFF_U     790528u     // 5120x512 f32 (10485760 B)

__device__ __forceinline__ ushort_t f2bf(float f) {
  unsigned u = __float_as_uint(f);
  unsigned r = u + 0x7FFFu + ((u >> 16) & 1u);   // round-to-nearest-even
  return (ushort_t)(r >> 16);
}

union Frag32B { uint4 q[2]; v16bf v; };

__device__ __forceinline__ v16bf load_frag(const ushort_t* p) {
  Frag32B f;
  f.q[0] = *(const uint4*)(p);
  f.q[1] = *(const uint4*)(p + 8);
  return f.v;
}

// swizzle a Dv/Dq-channel index so each lane's A-fragment (two 8-half chunks)
// becomes one contiguous 32B run: within each 32-group,
// pos = (d&7) | (bit d4 -> 3) | (bit d3 -> 4)
__device__ __forceinline__ int swz(int d) {
  return (d & ~31) | (d & 7) | (((d >> 4) & 1) << 3) | (((d >> 3) & 1) << 4);
}

// ---------------------------------------------------------------- norms ----
__global__ void norm_kernel(const float* __restrict__ Vv, const float* __restrict__ gv,
                            const float* __restrict__ Vq, const float* __restrict__ gq,
                            const float* __restrict__ Vl, const float* __restrict__ gl,
                            float* __restrict__ scal, float* __restrict__ wl) {
  __shared__ float red[256];
  __shared__ float sinv;
  const int bid = blockIdx.x, tid = threadIdx.x;
  const float* x; const float* g; int n;
  if (bid == 0)      { x = Vv; g = gv; n = H_ * DV_; }
  else if (bid == 1) { x = Vq; g = gq; n = H_ * DQ_; }
  else               { x = Vl; g = gl; n = H_; }
  float s = 0.f;
  for (int i = tid; i < n; i += 256) { float t = x[i]; s += t * t; }
  red[tid] = s; __syncthreads();
  for (int off = 128; off > 0; off >>= 1) {
    if (tid < off) red[tid] += red[tid + off];
    __syncthreads();
  }
  if (tid == 0) { float inv = g[0] * rsqrtf(red[0]); scal[bid] = inv; sinv = inv; }
  __syncthreads();
  if (bid == 2) for (int i = tid; i < H_; i += 256) wl[i] = Vl[i] * sinv;
}

// ----------------------------------------------------- pack weights bf16 ---
// B-operand fragment layout: frag f = ntile*ksteps + kstep; lane l holds
// column n = l&15, k = (j>>3)*16 + (l>>4)*8 + (j&7) (j=0..15) -> table order.
__global__ void pack_kernel(const float* __restrict__ Vv, const float* __restrict__ Vq,
                            const float* __restrict__ scal,
                            ushort_t* __restrict__ wvT, ushort_t* __restrict__ wqT) {
  const int tid = blockIdx.x * 256 + threadIdx.x;
  if (tid < H_ * DV_) {                       // Wv: 32 ntiles x 8 ksteps
    const float sv = scal[0];
    int f = tid >> 9, idx = tid & 511, lane = idx >> 4, j = idx & 15;
    int h = (f >> 3) * 16 + (lane & 15);
    int d = (f & 7) * 32 + ((j >> 3) << 4) + (lane >> 4) * 8 + (j & 7);
    wvT[tid] = f2bf(Vv[h * DV_ + d] * sv);
  } else {                                    // Wq: 32 ntiles x 16 ksteps
    int t = tid - H_ * DV_;
    const float sq = scal[1];
    int f = t >> 9, idx = t & 511, lane = idx >> 4, j = idx & 15;
    int h = (f >> 4) * 16 + (lane & 15);
    int d = (f & 15) * 32 + ((j >> 3) << 4) + (lane >> 4) * 8 + (j & 7);
    wqT[t] = f2bf(Vq[h * DQ_ + d] * sq);
  }
}

// ------------------------------------- u[b,l,h] = relu(ctx.Wq^T+bq)*wl -----
__global__ __launch_bounds__(256, 1)
void qp_kernel(const float* __restrict__ ctx, const float* __restrict__ bq,
               const float* __restrict__ wl, const ushort_t* __restrict__ wqT,
               float* __restrict__ u) {
  __shared__ ushort_t sC[32 * PQ_];
  const int r0 = blockIdx.x * 32;             // 160 blocks x 32 rows = 5120
  const int tid = threadIdx.x;

  // stage 32 rows x 512 f32 -> swizzled bf16
#pragma unroll
  for (int it = 0; it < 16; ++it) {
    int i4 = tid + it * 256;                  // 4096 float4
    int row = i4 >> 7;
    int d = (i4 & 127) << 2;
    float4 x = *(const float4*)(ctx + (size_t)(r0 + row) * DQ_ + d);
    ushort_t* dst = &sC[row * PQ_ + swz(d)];
    dst[0] = f2bf(x.x); dst[1] = f2bf(x.y); dst[2] = f2bf(x.z); dst[3] = f2bf(x.w);
  }
  __syncthreads();

  const int wave = tid >> 5, lane = tid & 31;
  const int n16 = lane & 15, hi = lane >> 4;

#pragma unroll
  for (int mtile = 0; mtile < 2; ++mtile) {
    v8f c[4] = {v8f{}, v8f{}, v8f{}, v8f{}};
    const ushort_t* abase = &sC[(mtile * 16 + n16) * PQ_];
    // unroll capped at 2: ~8 B-fragments in flight, no VGPR spill
#pragma unroll 2
    for (int kstep = 0; kstep < 16; ++kstep) {
      v16bf a = load_frag(abase + kstep * 32 + hi * 16);
#pragma unroll
      for (int i = 0; i < 4; ++i) {
        int nt = wave + 8 * i;
        v16bf b = load_frag(wqT + (size_t)(nt * 16 + kstep) * 512 + lane * 16);
        c[i] = __builtin_amdgcn_wmma_f32_16x16x32_bf16(
            false, a, false, b, (short)0, c[i], false, false);
      }
    }
#pragma unroll
    for (int i = 0; i < 4; ++i) {
      int nt = wave + 8 * i;
      int h = nt * 16 + n16;
      float bqh = bq[h], wlh = wl[h];
#pragma unroll
      for (int r = 0; r < 8; ++r) {
        int row = r0 + mtile * 16 + hi * 8 + r;
        u[(size_t)row * H_ + h] = fmaxf(c[i][r] + bqh, 0.f) * wlh;
      }
    }
  }
}

// --------------------------------------------------- fused attention core --
__global__ __launch_bounds__(256, 1)
void main_kernel(const float* __restrict__ v, const int* __restrict__ lengths,
                 const float* __restrict__ bv, const float* __restrict__ bl,
                 const float* __restrict__ u, const ushort_t* __restrict__ wvT,
                 float* __restrict__ out) {
  __shared__ ushort_t sV[KP_ * PV_];
  __shared__ float sU[H_];
  __shared__ float sLog[KP_];
  __shared__ float sW[KP_];

  const int p = blockIdx.x;                   // p = l*64 + b
  const int b = p & 63, l = p >> 6;
  const int tid = threadIdx.x;
  const int rr = b * L_ + l;

  if (tid < KP_) sLog[tid] = 0.f;
  const float* urow = u + (size_t)rr * H_;
#pragma unroll
  for (int it = 0; it < 2; ++it) sU[tid + it * 256] = urow[tid + it * 256];

  // stage v tile (36 x 256 f32) -> swizzled bf16
  const float* vt = v + (size_t)p * K_ * DV_;
#pragma unroll
  for (int it = 0; it < 9; ++it) {
    int i4 = tid + it * 256;                  // 2304 float4 = 36 rows x 64
    int row = i4 >> 6;
    int d = (i4 & 63) << 2;
    float4 x = *(const float4*)(vt + row * DV_ + d);
    ushort_t* dst = &sV[row * PV_ + swz(d)];
    dst[0] = f2bf(x.x); dst[1] = f2bf(x.y); dst[2] = f2bf(x.z); dst[3] = f2bf(x.w);
  }
  // zero pad rows 36..47
#pragma unroll
  for (int it = 0; it < 12; ++it) {
    int i = tid + it * 256;                   // 12 rows x 256 halves
    sV[(K_ + (i >> 8)) * PV_ + (i & 255)] = 0;
  }
  __syncthreads();

  const int wave = tid >> 5, lane = tid & 31;
  const int n16 = lane & 15, hi = lane >> 4;

#pragma unroll
  for (int mtile = 0; mtile < 3; ++mtile) {
    v8f c[4] = {v8f{}, v8f{}, v8f{}, v8f{}};
    const ushort_t* abase = &sV[(mtile * 16 + n16) * PV_];
    // unroll capped at 2: ~8 B-fragments in flight, no VGPR spill
#pragma unroll 2
    for (int kstep = 0; kstep < 8; ++kstep) {
      v16bf a = load_frag(abase + kstep * 32 + hi * 16);
#pragma unroll
      for (int i = 0; i < 4; ++i) {
        int nt = wave + 8 * i;
        v16bf bf = load_frag(wvT + (size_t)(nt * 8 + kstep) * 512 + lane * 16);
        c[i] = __builtin_amdgcn_wmma_f32_16x16x32_bf16(
            false, a, false, bf, (short)0, c[i], false, false);
      }
    }
    // fold: logits[row] += sum_h relu(c + bv[h]) * u[h]
    float acc[8] = {0, 0, 0, 0, 0, 0, 0, 0};
#pragma unroll
    for (int i = 0; i < 4; ++i) {
      int h = (wave + 8 * i) * 16 + n16;
      float uh = sU[h], bvh = bv[h];
#pragma unroll
      for (int r = 0; r < 8; ++r) acc[r] += fmaxf(c[i][r] + bvh, 0.f) * uh;
    }
#pragma unroll
    for (int r = 0; r < 8; ++r) {
      float s = acc[r];
      s += __shfl_xor(s, 1, 16);
      s += __shfl_xor(s, 2, 16);
      s += __shfl_xor(s, 4, 16);
      s += __shfl_xor(s, 8, 16);
      if (n16 == 0) atomicAdd(&sLog[mtile * 16 + hi * 8 + r], s);
    }
  }
  __syncthreads();

  // masked softmax over k (serial in one lane; K=36 is tiny)
  if (tid == 0) {
    const int len = lengths[p];
    const float bl0 = bl[0];
    float mx = -3.4e38f;
    for (int k = 0; k < K_; ++k) {
      float lg = (k < len) ? (sLog[k] + bl0) : NEGINF_;
      sLog[k] = lg;
      mx = fmaxf(mx, lg);
    }
    float s = 0.f;
    for (int k = 0; k < K_; ++k) { float e = __expf(sLog[k] - mx); sW[k] = e; s += e; }
    float inv = 1.f / s;
    for (int k = 0; k < KP_; ++k) sW[k] = (k < len) ? sW[k] * inv : 0.f;
  }
  __syncthreads();

  // out[b,l,d] = sum_k w[k] * v[l,b,k,d]  (f32 from global, coalesced)
  const int d = tid;
  float o = 0.f;
#pragma unroll 4
  for (int k = 0; k < K_; ++k) o = fmaf(sW[k], vt[k * DV_ + d], o);
  out[(size_t)rr * DV_ + d] = o;
}

// ---------------------------------------------------------------- launch ---
extern "C" void kernel_launch(void* const* d_in, const int* in_sizes, int n_in,
                              void* d_out, int out_size, void* d_ws, size_t ws_size,
                              hipStream_t stream) {
  const float* ctx     = (const float*)d_in[0];
  const float* v       = (const float*)d_in[1];
  const int*   lengths = (const int*)d_in[2];
  const float* Vv      = (const float*)d_in[3];
  const float* gv      = (const float*)d_in[4];
  const float* bv      = (const float*)d_in[5];
  const float* Vq      = (const float*)d_in[6];
  const float* gq      = (const float*)d_in[7];
  const float* bq      = (const float*)d_in[8];
  const float* Vl      = (const float*)d_in[9];
  const float* gl      = (const float*)d_in[10];
  const float* bl      = (const float*)d_in[11];
  float* out = (float*)d_out;

  char* ws = (char*)d_ws;
  float*    scal = (float*)(ws + OFF_SCAL);
  float*    wl   = (float*)(ws + OFF_WL);
  ushort_t* wvT  = (ushort_t*)(ws + OFF_WV);
  ushort_t* wqT  = (ushort_t*)(ws + OFF_WQ);
  float*    u    = (float*)(ws + OFF_U);

  norm_kernel<<<3, 256, 0, stream>>>(Vv, gv, Vq, gq, Vl, gl, scal, wl);
  pack_kernel<<<(H_ * DV_ + H_ * DQ_) / 256, 256, 0, stream>>>(Vv, Vq, scal, wvT, wqT);
  qp_kernel<<<(B_ * L_) / 32, 256, 0, stream>>>(ctx, bq, wl, wqT, u);
  main_kernel<<<L_ * B_, 256, 0, stream>>>(v, lengths, bv, bl, u, wvT, out);
}
